// SelfAttentiveBimodalFusion_40475771797773
// MI455X (gfx1250) — compile-verified, matrix operand order checked by reading, not run on hardware
//
#include <hip/hip_runtime.h>

typedef __attribute__((ext_vector_type(16))) _Float16 v16h;
typedef __attribute__((ext_vector_type(8)))  _Float16 v8h;
typedef __attribute__((ext_vector_type(8)))  float    v8f;

#define N_PTS 16384
#define NBLK  64          // N_PTS / 256 for row-parallel kernels
#define EPSBN 1e-5f
#define SLOPE 0.2f

// ---------------------------------------------------------------- reductions
__device__ __forceinline__ float wave_sum32(float x) {
#pragma unroll
  for (int m = 16; m >= 1; m >>= 1) x += __shfl_xor(x, m, 32);
  return x;
}

// lane <-> lane^16 exchange via v_permlanex16_b32 (VALU, no DS, no wait)
__device__ __forceinline__ float xor16(float x) {
  const int i = __float_as_int(x);
  const int r = __builtin_amdgcn_permlanex16(i, i, 0x76543210, 0xfedcba98, false, false);
  return __int_as_float(r);
}

// raw v_exp_f32 (flush-to-zero below -126 is fine for softmax tails)
__device__ __forceinline__ float fexp2(float x) { return __builtin_amdgcn_exp2f(x); }

// ------------------------------------------------- layer 1: concat -> @W1, stats
__global__ __launch_bounds__(256) void proj1_kernel(
    const float* __restrict__ x_main, const float* __restrict__ x_mod,
    const float* __restrict__ W1, float* __restrict__ H1,
    float* __restrict__ psum, float* __restrict__ psq) {
  __shared__ float W1s[192 * 16];
  __shared__ float wsum[8][16];
  __shared__ float wsq[8][16];
  for (int i = threadIdx.x; i < 192 * 16; i += 256) W1s[i] = W1[i];
  __syncthreads();

  const int r    = blockIdx.x * 256 + threadIdx.x;
  const int lane = threadIdx.x & 31;
  const int wave = threadIdx.x >> 5;

  float acc[16];
#pragma unroll
  for (int j = 0; j < 16; j++) acc[j] = 0.0f;

  const float* xr = x_main + (size_t)r * 64;
  for (int i = 0; i < 64; i++) {
    const float x = xr[i];
#pragma unroll
    for (int j = 0; j < 16; j++) acc[j] += x * W1s[i * 16 + j];
  }
  const float* xr2 = x_mod + (size_t)r * 128;
  for (int i = 0; i < 128; i++) {
    const float x = xr2[i];
#pragma unroll
    for (int j = 0; j < 16; j++) acc[j] += x * W1s[(64 + i) * 16 + j];
  }
#pragma unroll
  for (int j = 0; j < 16; j++) H1[(size_t)r * 16 + j] = acc[j];

#pragma unroll
  for (int j = 0; j < 16; j++) {
    const float s = wave_sum32(acc[j]);
    const float q = wave_sum32(acc[j] * acc[j]);
    if (lane == 0) { wsum[wave][j] = s; wsq[wave][j] = q; }
  }
  __syncthreads();
  if (threadIdx.x < 16) {
    float s = 0.0f, q = 0.0f;
    for (int w = 0; w < 8; w++) { s += wsum[w][threadIdx.x]; q += wsq[w][threadIdx.x]; }
    psum[blockIdx.x * 16 + threadIdx.x] = s;
    psq [blockIdx.x * 16 + threadIdx.x] = q;
  }
}

// ------------------------------------------ finalize BN affine: a = g*rstd, c = b - mean*a
__global__ void finalize_kernel(const float* __restrict__ psum, const float* __restrict__ psq,
                                const float* __restrict__ g, const float* __restrict__ b,
                                float* __restrict__ ab) {
  const int j = threadIdx.x;
  if (j < 16) {
    float s = 0.0f, q = 0.0f;
    for (int blk = 0; blk < NBLK; blk++) { s += psum[blk * 16 + j]; q += psq[blk * 16 + j]; }
    const float mean = s / (float)N_PTS;
    const float var  = q / (float)N_PTS - mean * mean;
    const float a    = g[j] * rsqrtf(var + EPSBN);
    ab[j]      = a;
    ab[16 + j] = b[j] - mean * a;
  }
}

// ------------------------------------------------- layer 2: BN1+LReLU -> @W2, stats
__global__ __launch_bounds__(256) void proj2_kernel(
    const float* __restrict__ H1, const float* __restrict__ ab1,
    const float* __restrict__ W2, float* __restrict__ H2,
    float* __restrict__ psum, float* __restrict__ psq) {
  __shared__ float W2s[16 * 16];
  __shared__ float abs_[32];
  __shared__ float wsum[8][16];
  __shared__ float wsq[8][16];
  if (threadIdx.x < 16 * 16) W2s[threadIdx.x] = W2[threadIdx.x];
  if (threadIdx.x < 32) abs_[threadIdx.x] = ab1[threadIdx.x];
  __syncthreads();

  const int r    = blockIdx.x * 256 + threadIdx.x;
  const int lane = threadIdx.x & 31;
  const int wave = threadIdx.x >> 5;

  float x1[16];
#pragma unroll
  for (int j = 0; j < 16; j++) {
    const float h = H1[(size_t)r * 16 + j] * abs_[j] + abs_[16 + j];
    x1[j] = (h >= 0.0f) ? h : SLOPE * h;
  }
  float acc[16];
#pragma unroll
  for (int j = 0; j < 16; j++) acc[j] = 0.0f;
#pragma unroll
  for (int i = 0; i < 16; i++) {
    const float x = x1[i];
#pragma unroll
    for (int j = 0; j < 16; j++) acc[j] += x * W2s[i * 16 + j];
  }
#pragma unroll
  for (int j = 0; j < 16; j++) H2[(size_t)r * 16 + j] = acc[j];

#pragma unroll
  for (int j = 0; j < 16; j++) {
    const float s = wave_sum32(acc[j]);
    const float q = wave_sum32(acc[j] * acc[j]);
    if (lane == 0) { wsum[wave][j] = s; wsq[wave][j] = q; }
  }
  __syncthreads();
  if (threadIdx.x < 16) {
    float s = 0.0f, q = 0.0f;
    for (int w = 0; w < 8; w++) { s += wsum[w][threadIdx.x]; q += wsq[w][threadIdx.x]; }
    psum[blockIdx.x * 16 + threadIdx.x] = s;
    psq [blockIdx.x * 16 + threadIdx.x] = q;
  }
}

// ------------------------------- BN2+LReLU -> Q(*scale*log2e) f16, K f16, V^T f16
__global__ __launch_bounds__(256) void proj3_kernel(
    const float* __restrict__ H2, const float* __restrict__ ab2,
    const float* __restrict__ Wq, const float* __restrict__ Wk, const float* __restrict__ Wv,
    _Float16* __restrict__ Qh, _Float16* __restrict__ Kh, _Float16* __restrict__ VT) {
  __shared__ float Wqs[16 * 8];
  __shared__ float Wks[16 * 8];
  __shared__ float Wvs[16 * 64];
  __shared__ float abs_[32];
  for (int i = threadIdx.x; i < 16 * 8; i += 256) { Wqs[i] = Wq[i]; Wks[i] = Wk[i]; }
  for (int i = threadIdx.x; i < 16 * 64; i += 256) Wvs[i] = Wv[i];
  if (threadIdx.x < 32) abs_[threadIdx.x] = ab2[threadIdx.x];
  __syncthreads();

  const int r = blockIdx.x * 256 + threadIdx.x;
  float x2[16];
#pragma unroll
  for (int j = 0; j < 16; j++) {
    const float h = H2[(size_t)r * 16 + j] * abs_[j] + abs_[16 + j];
    x2[j] = (h >= 0.0f) ? h : SLOPE * h;
  }
  // 1/sqrt(8) * log2(e): scores in log2 domain -> exp2 in softmax (no per-exp mul)
  const float qscale = 0.3535533905932738f * 1.4426950408889634f;
#pragma unroll
  for (int j = 0; j < 8; j++) {
    float q = 0.0f, k = 0.0f;
#pragma unroll
    for (int i = 0; i < 16; i++) { q += x2[i] * Wqs[i * 8 + j]; k += x2[i] * Wks[i * 8 + j]; }
    Qh[(size_t)r * 8 + j] = (_Float16)(q * qscale);
    Kh[(size_t)r * 8 + j] = (_Float16)k;
  }
#pragma unroll 4
  for (int c = 0; c < 64; c++) {
    float v = 0.0f;
#pragma unroll
    for (int i = 0; i < 16; i++) v += x2[i] * Wvs[i * 64 + c];
    VT[(size_t)c * N_PTS + r] = (_Float16)v;  // transposed store for A-operand loads
  }
}

// ---------------------------------------------------------------- flash attention
// one wave per 16-query tile; 64 keys per step.
// S^T = K . Q^T  (A = K-tile, B = Q^T): each lane owns one query column ->
// softmax is in-lane + one permlanex16 merge; exp2(S^T) lands directly in the
// B-operand layout for O^T = V^T . P^T. No LDS, no DS ops in the loop.
__global__ __launch_bounds__(256) void attn_kernel(
    const _Float16* __restrict__ Qh, const _Float16* __restrict__ Kh,
    const _Float16* __restrict__ VT, float* __restrict__ out) {
  const int lane    = threadIdx.x & 31;
  const int wave    = threadIdx.x >> 5;
  const int laneMod = lane & 15;
  const bool hi     = lane >= 16;
  const int qtile   = blockIdx.x * 8 + wave;
  const int qbase   = qtile * 16;

  // ---- Q^T B-operand: lane n owns query qbase+n, K(d)=0..7 in els 0-7; d=8..15 pad=0
  v16h qb = {};
  if (!hi) {
    const v8h qv = *(const v8h*)(Qh + (size_t)(qbase + laneMod) * 8);
#pragma unroll
    for (int e = 0; e < 8; e++) qb[e] = qv[e];
  }

  v8f o0 = {}, o1 = {}, o2 = {}, o3 = {};   // O^T accumulators, 4 chunks of 16 v-cols
  float m = -1e30f, l = 0.0f;               // per-query softmax state (lane-pair coherent)

  for (int kb = 0; kb < N_PTS; kb += 64) {
    // ---- K A-operands for 4 sub-tiles of 16 keys (rows = keys, K(d)=0..7, zero-pad)
    v16h ka0 = {}, ka1 = {}, ka2 = {}, ka3 = {};
    if (!hi) {
      const _Float16* kp = Kh + (size_t)(kb + laneMod) * 8;
      const v8h k0 = *(const v8h*)(kp);
      const v8h k1 = *(const v8h*)(kp + 16 * 8);
      const v8h k2 = *(const v8h*)(kp + 32 * 8);
      const v8h k3 = *(const v8h*)(kp + 48 * 8);
#pragma unroll
      for (int e = 0; e < 8; e++) { ka0[e] = k0[e]; ka1[e] = k1[e]; ka2[e] = k2[e]; ka3[e] = k3[e]; }
    }
    const v8f zc = {};
    // lane n slots: s0 -> keys kb+0..7, s1 -> +16..23, s2 -> +32..39, s3 -> +48..55
    // (lane n+16 holds the +8 variants)
    v8f s0 = __builtin_amdgcn_wmma_f32_16x16x32_f16(false, ka0, false, qb, (short)0, zc, false, false);
    v8f s1 = __builtin_amdgcn_wmma_f32_16x16x32_f16(false, ka1, false, qb, (short)0, zc, false, false);
    v8f s2 = __builtin_amdgcn_wmma_f32_16x16x32_f16(false, ka2, false, qb, (short)0, zc, false, false);
    v8f s3 = __builtin_amdgcn_wmma_f32_16x16x32_f16(false, ka3, false, qb, (short)0, zc, false, false);

    // ---- in-lane softmax over this lane's 32 keys, one cross-half merge
    float mx = fmaxf(fmaxf(s0[0], s1[0]), fmaxf(s2[0], s3[0]));
#pragma unroll
    for (int e = 1; e < 8; e++)
      mx = fmaxf(mx, fmaxf(fmaxf(s0[e], s1[e]), fmaxf(s2[e], s3[e])));
    mx = fmaxf(mx, xor16(mx));
    const float mn    = fmaxf(m, mx);
    const float alpha = fexp2(m - mn);
    m = mn;

    float p0[8], p1[8], p2[8], p3[8], rs = 0.0f;
#pragma unroll
    for (int e = 0; e < 8; e++) {
      p0[e] = fexp2(s0[e] - mn);
      p1[e] = fexp2(s1[e] - mn);
      p2[e] = fexp2(s2[e] - mn);
      p3[e] = fexp2(s3[e] - mn);
      rs += (p0[e] + p1[e]) + (p2[e] + p3[e]);
    }
    rs += xor16(rs);
    l = l * alpha + rs;

    o0 *= alpha; o1 *= alpha; o2 *= alpha; o3 *= alpha;

    // ---- P^T B-operands: this lane's values, no cross-lane movement
    v16h pb0, pb1;
#pragma unroll
    for (int e = 0; e < 8; e++) {
      pb0[e] = (_Float16)p0[e]; pb0[e + 8] = (_Float16)p1[e];
      pb1[e] = (_Float16)p2[e]; pb1[e + 8] = (_Float16)p3[e];
    }

    // ---- O^T += V^T . P^T  (V^T chunks as A-operands; contiguous b128 loads)
#pragma unroll
    for (int c = 0; c < 4; c++) {
      const _Float16* vp = VT + (size_t)(c * 16 + laneMod) * N_PTS + kb + (hi ? 8 : 0);
      const v8h v0 = *(const v8h*)(vp);
      const v8h v1 = *(const v8h*)(vp + 16);
      const v8h v2 = *(const v8h*)(vp + 32);
      const v8h v3 = *(const v8h*)(vp + 48);
      v16h va0, va1;
#pragma unroll
      for (int e = 0; e < 8; e++) {
        va0[e] = v0[e]; va0[e + 8] = v1[e];
        va1[e] = v2[e]; va1[e + 8] = v3[e];
      }
      v8f* oc = (c == 0) ? &o0 : (c == 1) ? &o1 : (c == 2) ? &o2 : &o3;
      *oc = __builtin_amdgcn_wmma_f32_16x16x32_f16(false, va0, false, pb0, (short)0, *oc, false, false);
      *oc = __builtin_amdgcn_wmma_f32_16x16x32_f16(false, va1, false, pb1, (short)0, *oc, false, false);
    }
  }

  // ---- epilogue: lane owns query qbase+laneMod; slot j = v-col c*16 + 8*hi + j
  const float invl = 1.0f / l;
  float* orow = out + (size_t)(qbase + laneMod) * 64 + (hi ? 8 : 0);
#pragma unroll
  for (int c = 0; c < 4; c++) {
    const v8f oc = (c == 0) ? o0 : (c == 1) ? o1 : (c == 2) ? o2 : o3;
    v8f sc = oc * invl;
    *(v8f*)(orow + c * 16) = sc;   // 32B aligned: 8 consecutive columns
  }
}

// ---------------------------------------------------------------- launcher
extern "C" void kernel_launch(void* const* d_in, const int* in_sizes, int n_in,
                              void* d_out, int out_size, void* d_ws, size_t ws_size,
                              hipStream_t stream) {
  (void)in_sizes; (void)n_in; (void)out_size; (void)ws_size;
  const float* x_main = (const float*)d_in[0];
  const float* x_mod  = (const float*)d_in[1];
  // d_in[2] = xyz (unused by reference math)
  const float* W1 = (const float*)d_in[3];
  const float* g1 = (const float*)d_in[4];
  const float* b1 = (const float*)d_in[5];
  const float* W2 = (const float*)d_in[6];
  const float* g2 = (const float*)d_in[7];
  const float* b2 = (const float*)d_in[8];
  const float* Wq = (const float*)d_in[9];
  const float* Wk = (const float*)d_in[10];
  const float* Wv = (const float*)d_in[11];

  char* ws = (char*)d_ws;
  float* H1 = (float*)ws;                                   // N*16 f32
  float* H2 = (float*)(ws + (size_t)N_PTS * 16 * 4);        // N*16 f32
  float* stats = (float*)(ws + (size_t)2 * N_PTS * 16 * 4); // partials + affine
  float* p1s = stats;            // 64*16
  float* p1q = stats + 1024;     // 64*16
  float* p2s = stats + 2048;     // 64*16
  float* p2q = stats + 3072;     // 64*16
  float* ab1 = stats + 4096;     // 32
  float* ab2 = stats + 4128;     // 32
  char* b2p = ws + (size_t)2 * N_PTS * 16 * 4 + 32768;
  _Float16* Qh = (_Float16*)b2p;                                  // N*8 f16 (scaled by 1/sqrt(8)*log2e)
  _Float16* Kh = (_Float16*)(b2p + (size_t)N_PTS * 8 * 2);        // N*8 f16
  _Float16* VT = (_Float16*)(b2p + (size_t)2 * N_PTS * 8 * 2);    // 64*N f16 (transposed)

  proj1_kernel<<<NBLK, 256, 0, stream>>>(x_main, x_mod, W1, H1, p1s, p1q);
  finalize_kernel<<<1, 16, 0, stream>>>(p1s, p1q, g1, b1, ab1);
  proj2_kernel<<<NBLK, 256, 0, stream>>>(H1, ab1, W2, H2, p2s, p2q);
  finalize_kernel<<<1, 16, 0, stream>>>(p2s, p2q, g2, b2, ab2);
  proj3_kernel<<<NBLK, 256, 0, stream>>>(H2, ab2, Wq, Wk, Wv, Qh, Kh, VT);
  attn_kernel<<<N_PTS / 16 / 8, 256, 0, stream>>>(Qh, Kh, VT, (float*)d_out);
}
